// Graph_Attention_Layer_31903017074983
// MI455X (gfx1250) — matrix operand order repaired and checked by reading, not compiled, not run on hardware
//
#include <hip/hip_runtime.h>
#include <hip/hip_bf16.h>

typedef __attribute__((ext_vector_type(16))) _Float16 v16h;
typedef __attribute__((ext_vector_type(8)))  _Float16 v8h;
typedef __attribute__((ext_vector_type(8)))  float    v8f;
typedef __attribute__((ext_vector_type(4)))  float    v4f;

#define N_GENES   16384
#define N_REG     8192
#define D_MODEL   256
#define OUT_DIM   128
#define ALPHA     0.2f
#define NEG_INF   (-9.0e15f)
#define L2E       1.4426950408889634f    // log2(e)

// ---------------------------------------------------------------------------
// Kernel 1: w_h = input @ weights  (f16 WMMA),
//           epilogue writes w_h as f16 COLUMN-MAJOR (whT[c*16384 + row])
//           and computes w_h1 = w_h[:8192] @ a[:128], w_h2 = w_h @ a[128:]
//           via in-register cross-lane reductions.
// One wave computes a 16x128 output tile. Block = 4 waves. Grid = 256 blocks.
// ---------------------------------------------------------------------------
__global__ void __launch_bounds__(128)
wh_kernel(const float* __restrict__ input,     // [16384,256]
          const float* __restrict__ weights,   // [256,128]
          const float* __restrict__ a_values,  // [256]
          _Float16* __restrict__ whT,          // [128,16384] col-major f16
          float* __restrict__ w_h1,            // [8192]
          float* __restrict__ w_h2)            // [16384]
{
    __shared__ _Float16 lds_w[OUT_DIM * D_MODEL];   // weights^T as f16: [c][k], 64 KB

    const int tid  = threadIdx.x;
    const int lane = tid & 31;
    const int wave = tid >> 5;
    const int h    = lane >> 4;      // K-group (0/1)
    const int r16  = lane & 15;
    const int m0   = (blockIdx.x * 4 + wave) * 16;

    // Cooperative fill of weights^T (f16) into LDS: coalesced global reads.
    for (int idx = tid; idx < OUT_DIM * D_MODEL; idx += 128) {
        const int c = idx & (OUT_DIM - 1);
        const int k = idx >> 7;
        lds_w[c * D_MODEL + k] = (_Float16)weights[k * OUT_DIM + c];
    }
    __syncthreads();

    v8f acc[8] = {};

    for (int k0 = 0; k0 < D_MODEL; k0 += 32) {
        // A fragment: row = m0 + r16; K = {h*16+0..15}
        const float* arow = input + (size_t)(m0 + r16) * D_MODEL + k0;
        v4f a0 = *(const v4f*)(arow + h * 16);
        v4f a1 = *(const v4f*)(arow + h * 16 + 4);
        v4f a2 = *(const v4f*)(arow + h * 16 + 8);
        v4f a3 = *(const v4f*)(arow + h * 16 + 12);
        v16h a;
#pragma unroll
        for (int t = 0; t < 4; ++t) {
            a[t]      = (_Float16)a0[t];
            a[4 + t]  = (_Float16)a1[t];
            a[8 + t]  = (_Float16)a2[t];
            a[12 + t] = (_Float16)a3[t];
        }
#pragma unroll
        for (int nt = 0; nt < 8; ++nt) {
            const int c = nt * 16 + r16;
            // B fragment: 16 consecutive K at column c, K-group h
            v16h b = *(const v16h*)(lds_w + c * D_MODEL + k0 + h * 16);
            acc[nt] = __builtin_amdgcn_wmma_f32_16x16x32_f16(
                false, a, false, b, (short)0, acc[nt], false, false);
        }
    }

    // --- epilogue 1: store w_h as f16 column-major ---
    // acc[nt][r] holds (row = m0 + h*8 + r, col = nt*16 + r16): 8 consecutive
    // rows per lane -> one 16B store per tile.
#pragma unroll
    for (int nt = 0; nt < 8; ++nt) {
        const int c = nt * 16 + r16;
        v8h p;
#pragma unroll
        for (int r = 0; r < 8; ++r) p[r] = (_Float16)acc[nt][r];
        *(v8h*)(whT + (size_t)c * N_GENES + m0 + h * 8) = p;
    }

    // --- epilogue 2: w_h1 / w_h2 row dots with a_values ---
    float p1[8] = {}, p2[8] = {};
#pragma unroll
    for (int nt = 0; nt < 8; ++nt) {
        const int c = nt * 16 + r16;
        const float av1 = a_values[c];
        const float av2 = a_values[OUT_DIM + c];
#pragma unroll
        for (int r = 0; r < 8; ++r) {
            p1[r] += acc[nt][r] * av1;
            p2[r] += acc[nt][r] * av2;
        }
    }
#pragma unroll
    for (int r = 0; r < 8; ++r) {
#pragma unroll
        for (int s = 1; s < 16; s <<= 1) {
            p1[r] += __shfl_xor(p1[r], s, 32);
            p2[r] += __shfl_xor(p2[r], s, 32);
        }
        if (r16 == 0) {
            const int row = m0 + h * 8 + r;
            w_h2[row] = p2[r];
            if (row < N_REG) w_h1[row] = p1[r];
        }
    }
}

// ---------------------------------------------------------------------------
// Kernel 2: w2max = max_j w_h2[j]  (single block reduction).
// leaky_relu is monotone increasing, so the unmasked row max is analytically
//   M_i = leaky_relu(w_h1[i] + w2max)
// which upper-bounds the masked row max; softmax is shift-invariant, so using
// M_i as the stabilizing shift is exact and removes ALL online-softmax state
// from the streaming kernel.
// ---------------------------------------------------------------------------
__global__ void __launch_bounds__(256)
w2max_kernel(const float* __restrict__ w_h2, float* __restrict__ w2max)
{
    __shared__ float red[256];
    float m = -3.0e38f;
    for (int j = threadIdx.x; j < N_GENES; j += 256)
        m = fmaxf(m, w_h2[j]);
    red[threadIdx.x] = m;
    __syncthreads();
    for (int s = 128; s > 0; s >>= 1) {
        if (threadIdx.x < (unsigned)s)
            red[threadIdx.x] = fmaxf(red[threadIdx.x], red[threadIdx.x + s]);
        __syncthreads();
    }
    if (threadIdx.x == 0) *w2max = red[0];
}

// ---------------------------------------------------------------------------
// Kernel 3: fused masked-attention softmax + att@w_h, single streaming pass.
// One wave handles 16 regulon rows, j in chunks of 32.
//  - shift M_i precomputed analytically (no online-softmax state)
//  - softmax denominator computed ON THE MATRIX UNIT: a 9th WMMA per chunk
//    with an all-ones B fragment accumulates sum_j p_ij into accS; the C/D
//    layout drops each row's sum into the same lane/register slot as its
//    output rows, so the epilogue needs no cross-lane reduction at all.
//  - adjacency mask applied as a multiply (adj is exactly 0.0/1.0)
// Block = 4 waves. Grid = 128 blocks.
// ---------------------------------------------------------------------------
__global__ void __launch_bounds__(128)
gat_kernel(const float* __restrict__ adj,      // [8192,16384]
           const _Float16* __restrict__ whT,   // [128,16384] col-major
           const float* __restrict__ w_h1,     // [8192]
           const float* __restrict__ w_h2,     // [16384]
           const float* __restrict__ w2maxp,   // [1]
           float* __restrict__ out)            // [8192,128]
{
    const int lane = threadIdx.x & 31;
    const int wave = threadIdx.x >> 5;
    const int h    = lane >> 4;
    const int r16  = lane & 15;
    const int i0   = (blockIdx.x * 4 + wave) * 16;

    const float w1  = w_h1[i0 + r16];          // this lane's A-row score bias
    const float w2m = *w2maxp;
    // analytic row-max bound: M = leaky_relu(w1 + w2max); shift in log2 domain
    const float Mx      = w1 + w2m;
    const float Mrow    = fmaxf(Mx, ALPHA * Mx);
    const float nmshift = -Mrow * L2E;         // p = exp2(e*L2E + nmshift)

    const float* adjrow = adj + (size_t)(i0 + r16) * N_GENES;

    v8f acc[8] = {};
    v8f accS   = {};                           // row sum-exp via P @ ones
    v16h bones;
#pragma unroll
    for (int t = 0; t < 16; ++t) bones[t] = (_Float16)1.0f;

    for (int j0 = 0; j0 < N_GENES; j0 += 32) {
        __builtin_prefetch(adjrow + j0 + 64, 0, 1);   // global_prefetch_b8

        // This lane's 16 (adj, w2) values in the A-fragment K pattern:
        // K = {h*16+0..15}
        v4f ad0 = *(const v4f*)(adjrow + j0 + h * 16);
        v4f ad1 = *(const v4f*)(adjrow + j0 + h * 16 + 4);
        v4f ad2 = *(const v4f*)(adjrow + j0 + h * 16 + 8);
        v4f ad3 = *(const v4f*)(adjrow + j0 + h * 16 + 12);
        v4f wa0 = *(const v4f*)(w_h2 + j0 + h * 16);
        v4f wa1 = *(const v4f*)(w_h2 + j0 + h * 16 + 4);
        v4f wa2 = *(const v4f*)(w_h2 + j0 + h * 16 + 8);
        v4f wa3 = *(const v4f*)(w_h2 + j0 + h * 16 + 12);

        v16h a;
#pragma unroll
        for (int t = 0; t < 4; ++t) {
            float e, arg, p;
            e = w1 + wa0[t]; e = fmaxf(e, ALPHA * e);
            arg = (e != 0.0f) ? __builtin_fmaf(e, L2E, nmshift) : -1.0e38f;
            p = __builtin_amdgcn_exp2f(arg) * ad0[t];      // adj mask: exact 0/1
            a[t] = (_Float16)p;

            e = w1 + wa1[t]; e = fmaxf(e, ALPHA * e);
            arg = (e != 0.0f) ? __builtin_fmaf(e, L2E, nmshift) : -1.0e38f;
            p = __builtin_amdgcn_exp2f(arg) * ad1[t];
            a[4 + t] = (_Float16)p;

            e = w1 + wa2[t]; e = fmaxf(e, ALPHA * e);
            arg = (e != 0.0f) ? __builtin_fmaf(e, L2E, nmshift) : -1.0e38f;
            p = __builtin_amdgcn_exp2f(arg) * ad2[t];
            a[8 + t] = (_Float16)p;

            e = w1 + wa3[t]; e = fmaxf(e, ALPHA * e);
            arg = (e != 0.0f) ? __builtin_fmaf(e, L2E, nmshift) : -1.0e38f;
            p = __builtin_amdgcn_exp2f(arg) * ad3[t];
            a[12 + t] = (_Float16)p;
        }

        // denominator: accS += P @ ones (row sums land in-lane, in-slot)
        accS = __builtin_amdgcn_wmma_f32_16x16x32_f16(
            false, a, false, bones, (short)0, accS, false, false);

        // acc += P (16x32) @ w_h[j0:j0+32, :]  over 8 column tiles
#pragma unroll
        for (int nt = 0; nt < 8; ++nt) {
            const _Float16* bp =
                whT + (size_t)(nt * 16 + r16) * N_GENES + j0 + h * 16;
            v16h b = *(const v16h*)bp;
            acc[nt] = __builtin_amdgcn_wmma_f32_16x16x32_f16(
                false, a, false, b, (short)0, acc[nt], false, false);
        }
    }

    // epilogue: divide by softmax denominator, ELU, store.
    // accS[r] already holds the full sum-exp for row i0 + h*8 + r (all
    // columns of the ones-product are identical) -> no shuffles needed.
#pragma unroll
    for (int r = 0; r < 8; ++r) {
        const float inv = 1.0f / accS[r];
        const int   row = i0 + h * 8 + r;
#pragma unroll
        for (int nt = 0; nt < 8; ++nt) {
            float x = acc[nt][r] * inv;
            x = x > 0.0f ? x : (__expf(x) - 1.0f);   // ELU, alpha=1
            out[(size_t)row * OUT_DIM + nt * 16 + r16] = x;
        }
    }
}

// ---------------------------------------------------------------------------
extern "C" void kernel_launch(void* const* d_in, const int* in_sizes, int n_in,
                              void* d_out, int out_size, void* d_ws, size_t ws_size,
                              hipStream_t stream) {
    const float* input    = (const float*)d_in[0];   // [16384,256]
    const float* adj      = (const float*)d_in[1];   // [8192,16384]
    const float* weights  = (const float*)d_in[2];   // [256,128]
    const float* a_values = (const float*)d_in[3];   // [256,1]
    float* out = (float*)d_out;                      // [8192,128]

    char* ws = (char*)d_ws;
    _Float16* whT = (_Float16*)ws;                                   // 4 MB
    float* w_h1   = (float*)(ws + (size_t)OUT_DIM * N_GENES * 2);    // 32 KB
    float* w_h2   = w_h1 + N_REG;                                    // 64 KB
    float* w2max  = w_h2 + N_GENES;                                  // 4 B

    // Kernel 1: 16384 rows / 16 per wave / 4 waves per block = 256 blocks
    wh_kernel<<<256, 128, 0, stream>>>(input, weights, a_values, whT, w_h1, w_h2);

    // Kernel 2: global max of w_h2 (enables analytic softmax shift)
    w2max_kernel<<<1, 256, 0, stream>>>(w_h2, w2max);

    // Kernel 3: 8192 rows / 16 per wave / 4 waves per block = 128 blocks
    gat_kernel<<<128, 128, 0, stream>>>(adj, whT, w_h1, w_h2, w2max, out);
}
